// MultiHeadAttention_13898514169910
// MI455X (gfx1250) — matrix operand order repaired and checked by reading, not compiled, not run on hardware
//
#include <hip/hip_runtime.h>
#include <stdint.h>

#define DEV __device__ __forceinline__

typedef __attribute__((ext_vector_type(8)))  __bf16 v8bf;
typedef __attribute__((ext_vector_type(16))) __bf16 v16bf;
typedef __attribute__((ext_vector_type(8)))  float  v8f;
typedef int i32x4 __attribute__((vector_size(16)));

namespace {
constexpr int kB = 2;
constexpr int kN = 2048;
constexpr int kL = 2048;
constexpr int kD = 1024;
constexpr int kH = 16;
constexpr int kC = 64;
}

#if __has_builtin(__builtin_amdgcn_global_load_async_to_lds_b128)
#define HAS_ASYNC_LDS 1
#endif
#if __has_builtin(__builtin_amdgcn_permlane16)
#define HAS_PERMLANE16 1
#endif

// ---------------------------------------------------------------------------
// Async global -> LDS copy (CDNA5 GLOBAL_LOAD_ASYNC_TO_LDS_B128, ASYNCcnt)
// with a synchronous fallback if the builtin is not available.
// Signature on ROCm clang-22: (i32x4 as1*, i32x4 as3*, imm offset, imm cpol).
// ---------------------------------------------------------------------------
#ifdef HAS_ASYNC_LDS
DEV void g2lds_b128(const __bf16* gsrc, __bf16* ldst) {
  __builtin_amdgcn_global_load_async_to_lds_b128(
      (__attribute__((address_space(1))) i32x4*)gsrc,
      (__attribute__((address_space(3))) i32x4*)ldst, 0, 0);
}
DEV void wait_async_all() {
#if __has_builtin(__builtin_amdgcn_s_wait_asynccnt)
  __builtin_amdgcn_s_wait_asynccnt(0);
#else
  asm volatile("s_wait_asynccnt 0x0" ::: "memory");
#endif
}
#else
DEV void g2lds_b128(const __bf16* gsrc, __bf16* ldst) {
  *(v8bf*)ldst = *(const v8bf*)gsrc;
}
DEV void wait_async_all() {}
#endif

// ---------------------------------------------------------------------------
// 16-lane XOR butterfly via v_permlane16_b32 (pure VALU; no LDS round trip).
// Selector nibbles: lane i reads lane (i ^ m) within its 16-lane row.
// ---------------------------------------------------------------------------
DEV float xor_lane16(float x, unsigned sel0, unsigned sel1, int xm) {
#ifdef HAS_PERMLANE16
  unsigned u = __builtin_bit_cast(unsigned, x);
  u = __builtin_amdgcn_permlane16(u, u, sel0, sel1, false, false);
  return __builtin_bit_cast(float, u);
#else
  return __shfl_xor(x, xm, 32);
#endif
}

DEV v8f row_reduce_max16(v8f v) {
  constexpr unsigned S0[4] = {0x67452301u, 0x54761032u, 0x32107654u, 0xFEDCBA98u};
  constexpr unsigned S1[4] = {0xEFCDAB89u, 0xDCFE98BAu, 0xBA98FEDCu, 0x76543210u};
#pragma unroll
  for (int s = 0; s < 4; ++s)
#pragma unroll
    for (int i = 0; i < 8; ++i)
      v[i] = fmaxf(v[i], xor_lane16(v[i], S0[s], S1[s], 1 << s));
  return v;
}

DEV v8f row_reduce_sum16(v8f v) {
  constexpr unsigned S0[4] = {0x67452301u, 0x54761032u, 0x32107654u, 0xFEDCBA98u};
  constexpr unsigned S1[4] = {0xEFCDAB89u, 0xDCFE98BAu, 0xBA98FEDCu, 0x76543210u};
#pragma unroll
  for (int s = 0; s < 4; ++s)
#pragma unroll
    for (int i = 0; i < 8; ++i)
      v[i] += xor_lane16(v[i], S0[s], S1[s], 1 << s);
  return v;
}

// ---------------------------------------------------------------------------
// WMMA helpers (CDNA5 gfx1250, wave32). Layouts per CDNA5 ISA:
//  A-frag (16x32 bf16): lane l<16 -> row l, K = {0..7, 16..23};
//                       lane l>=16 -> row l-16, K = {8..15, 24..31}
//  B-frag (32x16 bf16), tile stored column(n)-major with contiguous k:
//                       lane l<16 -> col l, K = {0..7, 16..23}; hi half -> K = {8..15, 24..31}
//  C/D (16x16 f32): lane l<16 vgpr r -> (M=r, N=l); hi half -> (M=8+r, N=l-16)
// ---------------------------------------------------------------------------
DEV v8f wmma_bf16(v16bf a, v16bf b, v8f c) {
  return __builtin_amdgcn_wmma_f32_16x16x32_bf16(
      /*neg_a=*/false, a, /*neg_b=*/false, b,
      /*c_mod=*/(short)0, c, /*reuse_a=*/false, /*reuse_b=*/false);
}

DEV v16bf load_frag_a(const __bf16* rowptr, int hi) {
  const __bf16* p = rowptr + hi * 8;
  v8bf lo = *(const v8bf*)(p);
  v8bf hv = *(const v8bf*)(p + 16);
  v16bf r;
#pragma unroll
  for (int i = 0; i < 8; ++i) { r[i] = lo[i]; r[i + 8] = hv[i]; }
  return r;
}

DEV v16bf load_frag_b(const __bf16* colptr, int hi) {
  const __bf16* p = colptr + hi * 8;
  v8bf lo = *(const v8bf*)(p);
  v8bf hv = *(const v8bf*)(p + 16);
  v16bf r;
#pragma unroll
  for (int i = 0; i < 8; ++i) { r[i] = lo[i]; r[i + 8] = hv[i]; }
  return r;
}

// ---------------------------------------------------------------------------
// Weight transpose + fp32->bf16 convert: Wt[n*D + k] = (bf16) W[k*D + n]
// ---------------------------------------------------------------------------
__global__ __launch_bounds__(256) void wt_convert_kernel(
    const float* __restrict__ W0, const float* __restrict__ W1,
    const float* __restrict__ W2, const float* __restrict__ W3,
    __bf16* __restrict__ O0, __bf16* __restrict__ O1,
    __bf16* __restrict__ O2, __bf16* __restrict__ O3) {
  const float* W;
  __bf16* O;
  switch (blockIdx.y) {
    case 0:  W = W0; O = O0; break;
    case 1:  W = W1; O = O1; break;
    case 2:  W = W2; O = O2; break;
    default: W = W3; O = O3; break;
  }
  const int idx = blockIdx.x * 256 + threadIdx.x;
  const int n = idx >> 10;
  const int k = idx & (kD - 1);
  O[(size_t)n * kD + k] = (__bf16)W[(size_t)k * kD + n];
}

// ---------------------------------------------------------------------------
// Generic GEMM: out(MxN) = A(MxK) * Wt(NxK)^T + bias, K = N = 1024.
// Block: 256 threads = 8 waves, tile 128x128, k-step 32.
// Wave grid 4(m) x 2(n); each wave computes 32x64 (8 wmma / k-step).
// B tile (and bf16 A tile) staged with async global->LDS DMA.
// ---------------------------------------------------------------------------
template <typename AT, bool OUT_BF16>
__global__ __launch_bounds__(256) void gemm_bias_kernel(
    const AT* __restrict__ A, const __bf16* __restrict__ Wt,
    const float* __restrict__ bias, void* __restrict__ out) {
  constexpr int K = kD, Nn = kD;
  constexpr int BM = 128, BN = 128, BK = 32;
  constexpr int LDT = BK + 8;  // 40 elems (80 B): 16B-aligned rows, bank skew
  __shared__ __bf16 sA[BM * LDT];
  __shared__ __bf16 sB[BN * LDT];

  const int bm = blockIdx.y * BM, bn = blockIdx.x * BN;
  const int tid = threadIdx.x;
  const int wave = tid >> 5, lane = tid & 31;
  const int lo16 = lane & 15, hi = lane >> 4;
  const int wm = (wave & 3) * 32, wn = (wave >> 2) * 64;

  v8f acc[2][4];
#pragma unroll
  for (int i = 0; i < 2; ++i)
#pragma unroll
    for (int j = 0; j < 4; ++j)
#pragma unroll
      for (int r = 0; r < 8; ++r) acc[i][j][r] = 0.0f;

  const int sr = tid >> 1;        // staging row (A) / col (B)
  const int sk = (tid & 1) * 16;  // staging k offset

  for (int k0 = 0; k0 < K; k0 += BK) {
    __syncthreads();
    {  // stage A tile
      const AT* src = A + (size_t)(bm + sr) * K + k0 + sk;
      __bf16* dst = sA + sr * LDT + sk;
      if constexpr (sizeof(AT) == 4) {
        // fp32 -> bf16 conversion has to go through VALU
        const float4* s4 = (const float4*)src;
        float4 x0 = s4[0], x1 = s4[1], x2 = s4[2], x3 = s4[3];
        v8bf c0 = v8bf{(__bf16)x0.x, (__bf16)x0.y, (__bf16)x0.z, (__bf16)x0.w,
                       (__bf16)x1.x, (__bf16)x1.y, (__bf16)x1.z, (__bf16)x1.w};
        v8bf c1 = v8bf{(__bf16)x2.x, (__bf16)x2.y, (__bf16)x2.z, (__bf16)x2.w,
                       (__bf16)x3.x, (__bf16)x3.y, (__bf16)x3.z, (__bf16)x3.w};
        ((v8bf*)dst)[0] = c0;
        ((v8bf*)dst)[1] = c1;
      } else {
        g2lds_b128((const __bf16*)src, dst);
        g2lds_b128((const __bf16*)src + 8, dst + 8);
      }
    }
    {  // stage B tile (bf16, n-major / k-contiguous): async DMA to LDS
      const __bf16* src = Wt + (size_t)(bn + sr) * K + k0 + sk;
      __bf16* dst = sB + sr * LDT + sk;
      g2lds_b128(src, dst);
      g2lds_b128(src + 8, dst + 8);
    }
    if (k0 + BK < K) {
      __builtin_prefetch(A + (size_t)(bm + sr) * K + k0 + BK + sk, 0, 0);
      __builtin_prefetch(Wt + (size_t)(bn + sr) * K + k0 + BK + sk, 0, 0);
    }
    wait_async_all();
    __syncthreads();

    v16bf af[2], bfr[4];
#pragma unroll
    for (int i = 0; i < 2; ++i)
      af[i] = load_frag_a(sA + (wm + i * 16 + lo16) * LDT, hi);
#pragma unroll
    for (int j = 0; j < 4; ++j)
      bfr[j] = load_frag_b(sB + (wn + j * 16 + lo16) * LDT, hi);
#pragma unroll
    for (int i = 0; i < 2; ++i)
#pragma unroll
      for (int j = 0; j < 4; ++j) acc[i][j] = wmma_bf16(af[i], bfr[j], acc[i][j]);
  }

  // epilogue: +bias, store (C/D layout: row = vgpr (+8 hi half), col = lane&15)
#pragma unroll
  for (int j = 0; j < 4; ++j) {
    const int gn = bn + wn + j * 16 + lo16;
    const float bv = bias[gn];
#pragma unroll
    for (int i = 0; i < 2; ++i) {
      const int gm0 = bm + wm + i * 16 + hi * 8;
#pragma unroll
      for (int r = 0; r < 8; ++r) {
        const float v = acc[i][j][r] + bv;
        const size_t off = (size_t)(gm0 + r) * Nn + gn;
        if constexpr (OUT_BF16)
          ((__bf16*)out)[off] = (__bf16)v;
        else
          ((float*)out)[off] = v;
      }
    }
  }
}

// ---------------------------------------------------------------------------
// Flash attention: block = (b, h, 128-query tile), 8 waves x 16 query rows.
// Per 32-key chunk: 4 wmma for S = Q K^T (k-dim = C = 64), masked scale,
// online softmax (permlane16 row reductions), P -> bf16 A-frag via a
// per-wave LDS round trip, 4 wmma for O += P V.  K staged by async DMA.
// ---------------------------------------------------------------------------
__global__ __launch_bounds__(256) void flash_attn_kernel(
    const __bf16* __restrict__ Qg, const __bf16* __restrict__ Kg,
    const __bf16* __restrict__ Vg, const uint8_t* __restrict__ mask,
    __bf16* __restrict__ Og) {
  constexpr int BKEY = 32;
  constexpr int LDK = 72;  // [32 keys][64 ch] rows, padded, 16B-aligned
  constexpr int LDV = 40;  // [64 ch][32 keys] rows (transposed), padded
  constexpr int LDP = 40;  // per-wave [16 rows][32 keys], padded
  __shared__ __bf16 sK[32 * LDK];
  __shared__ __bf16 sV[64 * LDV];
  __shared__ __bf16 sP[8 * 16 * LDP];

  const int b = blockIdx.z, h = blockIdx.y;
  const int tid = threadIdx.x, wave = tid >> 5, lane = tid & 31;
  const int lo16 = lane & 15, hi = lane >> 4;
  const int q0 = blockIdx.x * 128 + wave * 16;
  const size_t hoff = (size_t)h * kC;

  // Q fragments (held in registers for the whole pass)
  v16bf qf0, qf1;
  {
    const __bf16* qrow = Qg + ((size_t)b * kN + q0 + lo16) * kD + hoff;
    qf0 = load_frag_a(qrow, hi);
    qf1 = load_frag_a(qrow + 32, hi);
  }

  v8f oacc[4], mvec, lvec;
#pragma unroll
  for (int r = 0; r < 8; ++r) { mvec[r] = -3.0e38f; lvec[r] = 0.0f; }
#pragma unroll
  for (int t = 0; t < 4; ++t)
#pragma unroll
    for (int r = 0; r < 8; ++r) oacc[t][r] = 0.0f;

  const int str = tid >> 3;        // staging key row (0..31)
  const int stc = (tid & 7) * 8;   // staging channel (0..56)
  const float dp = 0.125f;         // C^-0.5
  const float L2E = 1.44269504f;
  const float NEG = -1.0e30f;

  for (int kc = 0; kc < kL; kc += BKEY) {
    __syncthreads();
    {  // stage K row-major via async DMA; V transposed through VALU
      const __bf16* ksrc = Kg + ((size_t)b * kL + kc + str) * kD + hoff + stc;
      g2lds_b128(ksrc, sK + str * LDK + stc);
      const __bf16* vsrc = Vg + ((size_t)b * kL + kc + str) * kD + hoff + stc;
      v8bf vv = *(const v8bf*)vsrc;
#pragma unroll
      for (int j = 0; j < 8; ++j) sV[(stc + j) * LDV + str] = vv[j];
    }
    wait_async_all();
    __syncthreads();

    // scores: two 16x16 tiles, k-dim = 64 channels
    v8f sc[2];
#pragma unroll
    for (int nt = 0; nt < 2; ++nt) {
      const __bf16* kp = sK + (nt * 16 + lo16) * LDK;
      v8f a;
#pragma unroll
      for (int r = 0; r < 8; ++r) a[r] = 0.0f;
      a = wmma_bf16(qf0, load_frag_b(kp, hi), a);
      a = wmma_bf16(qf1, load_frag_b(kp + 32, hi), a);
      const int keyg = kc + nt * 16 + lo16;
      const bool mk = mask[(size_t)b * kL + keyg] != 0;
#pragma unroll
      for (int r = 0; r < 8; ++r) sc[nt][r] = mk ? NEG : a[r] * dp;
    }

    // row stats over 32 keys (lanes hold columns; vgpr slot = row)
    v8f rmx;
#pragma unroll
    for (int r = 0; r < 8; ++r) rmx[r] = fmaxf(sc[0][r], sc[1][r]);
    rmx = row_reduce_max16(rmx);

    v8f mnew, alpha;
#pragma unroll
    for (int r = 0; r < 8; ++r) {
      mnew[r] = fmaxf(mvec[r], rmx[r]);
      alpha[r] = exp2f((mvec[r] - mnew[r]) * L2E);
    }
#pragma unroll
    for (int nt = 0; nt < 2; ++nt)
#pragma unroll
      for (int r = 0; r < 8; ++r)
        sc[nt][r] = exp2f((sc[nt][r] - mnew[r]) * L2E);

    v8f rs;
#pragma unroll
    for (int r = 0; r < 8; ++r) rs[r] = sc[0][r] + sc[1][r];
    rs = row_reduce_sum16(rs);

#pragma unroll
    for (int r = 0; r < 8; ++r) {
      lvec[r] = lvec[r] * alpha[r] + rs[r];
      mvec[r] = mnew[r];
    }
#pragma unroll
    for (int t = 0; t < 4; ++t)
#pragma unroll
      for (int r = 0; r < 8; ++r) oacc[t][r] *= alpha[r];

    // P (f32 C/D layout) -> bf16 A-frag via per-wave LDS region
    __bf16* pw = sP + wave * 16 * LDP;
#pragma unroll
    for (int nt = 0; nt < 2; ++nt)
#pragma unroll
      for (int r = 0; r < 8; ++r)
        pw[(hi * 8 + r) * LDP + nt * 16 + lo16] = (__bf16)sc[nt][r];
    v16bf pf = load_frag_a(pw + lo16 * LDP, hi);

    // O += P * V
#pragma unroll
    for (int t = 0; t < 4; ++t) {
      v16bf vf = load_frag_b(sV + (t * 16 + lo16) * LDV, hi);
      oacc[t] = wmma_bf16(pf, vf, oacc[t]);
    }
  }

  v8f inv;
#pragma unroll
  for (int r = 0; r < 8; ++r) inv[r] = 1.0f / lvec[r];
#pragma unroll
  for (int t = 0; t < 4; ++t)
#pragma unroll
    for (int r = 0; r < 8; ++r) {
      const int row = q0 + hi * 8 + r;
      const size_t off = ((size_t)b * kN + row) * kD + hoff + t * 16 + lo16;
      Og[off] = (__bf16)(oacc[t][r] * inv[r]);
    }
}

// ---------------------------------------------------------------------------
// Launcher
// ---------------------------------------------------------------------------
extern "C" void kernel_launch(void* const* d_in, const int* in_sizes, int n_in,
                              void* d_out, int out_size, void* d_ws, size_t ws_size,
                              hipStream_t stream) {
  const float* x_q = (const float*)d_in[0];
  const float* x_k = (const float*)d_in[1];
  const float* x_v = (const float*)d_in[2];
  const float* Wq = (const float*)d_in[3];
  const float* bq = (const float*)d_in[4];
  const float* Wk = (const float*)d_in[5];
  const float* bk = (const float*)d_in[6];
  const float* Wv = (const float*)d_in[7];
  const float* bv = (const float*)d_in[8];
  const float* Wo = (const float*)d_in[9];
  const float* bo = (const float*)d_in[10];
  const uint8_t* pad_mask = (const uint8_t*)d_in[11];
  float* out = (float*)d_out;

  char* ws = (char*)d_ws;
  const size_t wbytes = (size_t)kD * kD * sizeof(__bf16);       // 2 MB each
  const size_t xbytes = (size_t)kB * kN * kD * sizeof(__bf16);  // 8 MB each
  __bf16* Wtq = (__bf16*)(ws + 0 * wbytes);
  __bf16* Wtk = (__bf16*)(ws + 1 * wbytes);
  __bf16* Wtv = (__bf16*)(ws + 2 * wbytes);
  __bf16* Wto = (__bf16*)(ws + 3 * wbytes);
  __bf16* Qg  = (__bf16*)(ws + 4 * wbytes + 0 * xbytes);
  __bf16* Kg  = (__bf16*)(ws + 4 * wbytes + 1 * xbytes);
  __bf16* Vg  = (__bf16*)(ws + 4 * wbytes + 2 * xbytes);
  __bf16* Ag  = (__bf16*)(ws + 4 * wbytes + 3 * xbytes);

  dim3 tb(256);
  wt_convert_kernel<<<dim3((kD * kD) / 256, 4), tb, 0, stream>>>(
      Wq, Wk, Wv, Wo, Wtq, Wtk, Wtv, Wto);
  gemm_bias_kernel<float, true><<<dim3(kD / 128, (kB * kN) / 128), tb, 0, stream>>>(
      x_q, Wtq, bq, Qg);
  gemm_bias_kernel<float, true><<<dim3(kD / 128, (kB * kL) / 128), tb, 0, stream>>>(
      x_k, Wtk, bk, Kg);
  gemm_bias_kernel<float, true><<<dim3(kD / 128, (kB * kL) / 128), tb, 0, stream>>>(
      x_v, Wtv, bv, Vg);
  flash_attn_kernel<<<dim3(kN / 128, kH, kB), tb, 0, stream>>>(
      Qg, Kg, Vg, pad_mask, Ag);
  gemm_bias_kernel<__bf16, false><<<dim3(kD / 128, (kB * kN) / 128), tb, 0, stream>>>(
      Ag, Wto, bo, out);

  (void)in_sizes; (void)n_in; (void)out_size; (void)ws_size;
}
